// Attention_61856118997257
// MI455X (gfx1250) — compile-verified, matrix operand order checked by reading, not run on hardware
//
#include <hip/hip_runtime.h>

// ---------------------------------------------------------------------------
// Types / helpers
// ---------------------------------------------------------------------------
typedef float  v8f   __attribute__((ext_vector_type(8)));
typedef __bf16 v16bf __attribute__((ext_vector_type(16)));

union FragBF {                 // 16 bf16 = 32B = 8 VGPRs (A/B fragment)
    v16bf v;
    uint4 q[2];
    unsigned short h[16];
};
union FragF {                  // 8 f32 = 32B = 8 VGPRs (C/D fragment)
    v8f   v;
    float f[8];
};

__device__ __forceinline__ unsigned short f2bf(float f) {
    union { float f; unsigned u; } x; x.f = f;
    unsigned r = x.u + 0x7FFFu + ((x.u >> 16) & 1u);   // round-to-nearest-even
    return (unsigned short)(r >> 16);
}

__device__ __forceinline__ v8f zero_v8f() {
    v8f z;
#pragma unroll
    for (int i = 0; i < 8; ++i) z[i] = 0.0f;
    return z;
}

__device__ __forceinline__ v8f wmma_bf16(const FragBF& a, const FragBF& b, v8f c) {
    return __builtin_amdgcn_wmma_f32_16x16x32_bf16(
        /*neg_a=*/false, a.v, /*neg_b=*/false, b.v,
        /*c_mod=*/(short)0, c, /*reuse_a=*/false, /*reuse_b=*/false);
}

// Problem constants
#define Bsz 2
#define Ssz 2048
#define Dsz 1024
#define Hsz 16
#define HD  64

// ---------------------------------------------------------------------------
// fp32 -> bf16 convert
// ---------------------------------------------------------------------------
__global__ void cvt_bf16_kernel(const float* __restrict__ in,
                                unsigned short* __restrict__ out, int n) {
    int i = blockIdx.x * blockDim.x + threadIdx.x;
    if (i < n) out[i] = f2bf(in[i]);
}

// ---------------------------------------------------------------------------
// Fused QKV projection + RoPE. 32x64 wave tile (2 row fragments, B reused).
// grid = (M/32, H, 3[z:0=Q,1=K,2=V]), block = 32 (one wave).
// Q,K  -> bf16  [bh][s][64]   (RoPE applied)
// V    -> bf16  [bh][d][s]    (transposed for contiguous PV B-fragments)
// ---------------------------------------------------------------------------
__global__ __launch_bounds__(32)
void qkv_rope_kernel(const unsigned short* __restrict__ xb,
                     const unsigned short* __restrict__ wqb,
                     const unsigned short* __restrict__ wkb,
                     const unsigned short* __restrict__ wvb,
                     const float* __restrict__ bq,
                     const float* __restrict__ bv,
                     const float* __restrict__ cosp,
                     const float* __restrict__ sinp,
                     unsigned short* __restrict__ Qb,
                     unsigned short* __restrict__ Kb,
                     unsigned short* __restrict__ Vtb)
{
    const int lane   = threadIdx.x;
    const int half   = lane >> 4;      // 0: K-offsets {0..7,16..23}, 1: {8..15,24..31}
    const int ln     = lane & 15;
    const int m0     = blockIdx.x * 32;    // row tile in [0, B*S)
    const int h      = blockIdx.y;         // head
    const int which  = blockIdx.z;         // 0=Q 1=K 2=V
    const int kShift = half ? 8 : 0;

    const unsigned short* W = (which == 0) ? wqb : (which == 1) ? wkb : wvb;

    v8f acc[2][4];
#pragma unroll
    for (int r = 0; r < 2; ++r)
#pragma unroll
        for (int t = 0; t < 4; ++t) acc[r][t] = zero_v8f();

    const unsigned short* aBase0 = xb + (size_t)(m0 + ln) * Dsz + kShift;
    const unsigned short* aBase1 = aBase0 + 16 * Dsz;
    for (int k0 = 0; k0 < Dsz; k0 += 32) {
        __builtin_prefetch((const void*)(aBase0 + k0 + 128), 0, 1);
        __builtin_prefetch((const void*)(aBase1 + k0 + 128), 0, 1);
        FragBF a0, a1;
        a0.q[0] = *(const uint4*)(aBase0 + k0);
        a0.q[1] = *(const uint4*)(aBase0 + k0 + 16);
        a1.q[0] = *(const uint4*)(aBase1 + k0);
        a1.q[1] = *(const uint4*)(aBase1 + k0 + 16);
#pragma unroll
        for (int t = 0; t < 4; ++t) {
            const int col = h * HD + t * 16 + ln;      // output channel o (B-col)
            const unsigned short* bp = W + (size_t)col * Dsz + k0 + kShift;
            __builtin_prefetch((const void*)(bp + 128), 0, 1);
            FragBF b;
            b.q[0] = *(const uint4*)(bp);
            b.q[1] = *(const uint4*)(bp + 16);
            acc[0][t] = wmma_bf16(a0, b, acc[0][t]);
            acc[1][t] = wmma_bf16(a1, b, acc[1][t]);
        }
    }

    const int bIdx  = m0 >> 11;           // / S
    const int sBase = m0 & (Ssz - 1);
    const int bh    = bIdx * Hsz + h;

    if (which == 2) {                      // ---- V: bias + transposed store
#pragma unroll
        for (int r = 0; r < 2; ++r) {
#pragma unroll
            for (int t = 0; t < 4; ++t) {
                const int   j    = t * 16 + ln;
                const float bias = bv[h * HD + j];
                FragF c; c.v = acc[r][t];
#pragma unroll
                for (int mi = 0; mi < 8; ++mi) {
                    const int s = sBase + r * 16 + mi + 8 * half;
                    Vtb[((size_t)bh * HD + j) * Ssz + s] = f2bf(c.f[mi] + bias);
                }
            }
        }
    } else {                               // ---- Q/K: bias(Q) + RoPE + store
        unsigned short* dst = (which == 0) ? Qb : Kb;
#pragma unroll
        for (int r = 0; r < 2; ++r) {
            FragF c[4];
#pragma unroll
            for (int t = 0; t < 4; ++t) c[t].v = acc[r][t];
            if (which == 0) {
#pragma unroll
                for (int t = 0; t < 4; ++t) {
                    const float bias = bq[h * HD + t * 16 + ln];
#pragma unroll
                    for (int mi = 0; mi < 8; ++mi) c[t].f[mi] += bias;
                }
            }
#pragma unroll
            for (int t = 0; t < 4; ++t) {
                const int j = t * 16 + ln;             // position within head dim
#pragma unroll
                for (int mi = 0; mi < 8; ++mi) {
                    const int   s    = sBase + r * 16 + mi + 8 * half;
                    const float cs   = cosp[s * HD + j];
                    const float sn   = sinp[s * HD + j];
                    const float pair = (t < 2) ? c[t + 2].f[mi] : c[t - 2].f[mi];
                    const float val  = (t < 2) ? (c[t].f[mi] * cs - pair * sn)
                                               : (c[t].f[mi] * cs + pair * sn);
                    dst[((size_t)bh * Ssz + s) * HD + j] = f2bf(val);
                }
            }
        }
    }
}

// ---------------------------------------------------------------------------
// Causal flash attention. grid = (B*H, S/16), block = 32.
// One wave: 16-row q-tile, online softmax over 32-wide k-tiles.
// Writes normalized context as bf16 vec [b][s][D] for the output projection.
// ---------------------------------------------------------------------------
__global__ __launch_bounds__(32)
void attn_kernel(const unsigned short* __restrict__ Qb,
                 const unsigned short* __restrict__ Kb,
                 const unsigned short* __restrict__ Vtb,
                 unsigned short* __restrict__ vecb)
{
    __shared__ unsigned short Plds[16 * 32];   // P tile staging (row-major 16x32)

    const int lane   = threadIdx.x;
    const int half   = lane >> 4;
    const int ln     = lane & 15;
    const int bh     = blockIdx.x;             // b*H + h
    const int q0     = blockIdx.y * 16;
    const int kShift = half ? 8 : 0;

    // Q A-fragments for the two K=32 steps over HD=64
    FragBF qa[2];
    {
        const unsigned short* qp = Qb + ((size_t)bh * Ssz + q0 + ln) * HD + kShift;
        qa[0].q[0] = *(const uint4*)(qp);
        qa[0].q[1] = *(const uint4*)(qp + 16);
        qa[1].q[0] = *(const uint4*)(qp + 32);
        qa[1].q[1] = *(const uint4*)(qp + 48);
    }

    FragF oacc[4];
#pragma unroll
    for (int t = 0; t < 4; ++t) oacc[t].v = zero_v8f();
    float mrow[8], lrow[8];
#pragma unroll
    for (int i = 0; i < 8; ++i) { mrow[i] = -3.0e38f; lrow[i] = 0.0f; }

    const float scale = 0.125f;                 // 1/sqrt(HD)
    const int   ktmax = (q0 + 15) >> 5;

    for (int kt = 0; kt <= ktmax; ++kt) {
        const int k0 = kt << 5;

        // prefetch next K/V tiles (speculative; dropped if past end)
        {
            const unsigned short* kpn =
                Kb + ((size_t)bh * Ssz + k0 + 32 + ln) * HD + kShift;
            __builtin_prefetch((const void*)(kpn), 0, 1);
            const unsigned short* vpn =
                Vtb + ((size_t)bh * HD + ln) * Ssz + k0 + 32 + kShift;
            __builtin_prefetch((const void*)(vpn), 0, 1);
        }

        // ---- scores: S0 (cols k0..k0+15), S1 (cols k0+16..k0+31)
        v8f s0 = zero_v8f(), s1 = zero_v8f();
#pragma unroll
        for (int d = 0; d < 2; ++d) {
            const unsigned short* kp =
                Kb + ((size_t)bh * Ssz + k0 + ln) * HD + d * 32 + kShift;
            FragBF kb0, kb1;
            kb0.q[0] = *(const uint4*)(kp);
            kb0.q[1] = *(const uint4*)(kp + 16);
            kb1.q[0] = *(const uint4*)(kp + 16 * HD);
            kb1.q[1] = *(const uint4*)(kp + 16 * HD + 16);
            s0 = wmma_bf16(qa[d], kb0, s0);
            s1 = wmma_bf16(qa[d], kb1, s1);
        }

        // ---- scale + causal mask (only the diagonal tile needs masking)
        FragF f0, f1; f0.v = s0; f1.v = s1;
        const bool diag = (kt == ktmax);
#pragma unroll
        for (int mi = 0; mi < 8; ++mi) {
            const int qrow = q0 + mi + 8 * half;
            float a0 = f0.f[mi] * scale;
            float a1 = f1.f[mi] * scale;
            if (diag) {
                if (k0 + ln      > qrow) a0 = -3.0e38f;
                if (k0 + 16 + ln > qrow) a1 = -3.0e38f;
            }
            f0.f[mi] = a0; f1.f[mi] = a1;
        }

        // ---- online softmax (row = C-layout vgpr index; lane halves = row halves)
#pragma unroll
        for (int mi = 0; mi < 8; ++mi) {
            float tm = fmaxf(f0.f[mi], f1.f[mi]);
#pragma unroll
            for (int off = 1; off < 16; off <<= 1)
                tm = fmaxf(tm, __shfl_xor(tm, off, 32));
            const float mn    = fmaxf(mrow[mi], tm);
            const float alpha = __expf(mrow[mi] - mn);
            mrow[mi] = mn;
            const float p0 = __expf(f0.f[mi] - mn);
            const float p1 = __expf(f1.f[mi] - mn);
            f0.f[mi] = p0; f1.f[mi] = p1;
            float ps = p0 + p1;
#pragma unroll
            for (int off = 1; off < 16; off <<= 1)
                ps += __shfl_xor(ps, off, 32);
            lrow[mi] = lrow[mi] * alpha + ps;
#pragma unroll
            for (int t = 0; t < 4; ++t) oacc[t].f[mi] *= alpha;
        }

        // ---- P: C-layout -> LDS row-major -> A-fragment
#pragma unroll
        for (int mi = 0; mi < 8; ++mi) {
            const int row = mi + 8 * half;
            Plds[row * 32 + ln]      = f2bf(f0.f[mi]);
            Plds[row * 32 + 16 + ln] = f2bf(f1.f[mi]);
        }
        __syncthreads();                       // single-wave WG: compiler fence + dscnt
        FragBF pa;
        {
            const unsigned short* pp = Plds + ln * 32 + kShift;
            pa.q[0] = *(const uint4*)(pp);
            pa.q[1] = *(const uint4*)(pp + 16);
        }

        // ---- PV: accumulate 16x64 output (4 N-tiles of 16)
#pragma unroll
        for (int t = 0; t < 4; ++t) {
            const unsigned short* vp =
                Vtb + ((size_t)bh * HD + t * 16 + ln) * Ssz + k0 + kShift;
            FragBF vb;
            vb.q[0] = *(const uint4*)(vp);
            vb.q[1] = *(const uint4*)(vp + 16);
            oacc[t].v = wmma_bf16(pa, vb, oacc[t].v);
        }
        __syncthreads();
    }

    // ---- normalize + store context vec bf16 [b][s][D]
    const int b = bh >> 4, h = bh & 15;
#pragma unroll
    for (int mi = 0; mi < 8; ++mi) {
        const float inv = 1.0f / lrow[mi];
        const int   s   = q0 + mi + 8 * half;
#pragma unroll
        for (int t = 0; t < 4; ++t) {
            const int col = h * HD + t * 16 + ln;
            vecb[((size_t)b * Ssz + s) * Dsz + col] = f2bf(oacc[t].f[mi] * inv);
        }
    }
}

// ---------------------------------------------------------------------------
// Output projection: out = vec @ wo^T + bo (fp32 out). 32x64 wave tile.
// grid = (M/32, N/64).
// ---------------------------------------------------------------------------
__global__ __launch_bounds__(32)
void oproj_kernel(const unsigned short* __restrict__ vecb,
                  const unsigned short* __restrict__ wob,
                  const float* __restrict__ bo,
                  float* __restrict__ out)
{
    const int lane   = threadIdx.x;
    const int half   = lane >> 4;
    const int ln     = lane & 15;
    const int m0     = blockIdx.x * 32;
    const int n0     = blockIdx.y * 64;
    const int kShift = half ? 8 : 0;

    FragF acc[2][4];
#pragma unroll
    for (int r = 0; r < 2; ++r)
#pragma unroll
        for (int t = 0; t < 4; ++t) acc[r][t].v = zero_v8f();

    const unsigned short* aBase0 = vecb + (size_t)(m0 + ln) * Dsz + kShift;
    const unsigned short* aBase1 = aBase0 + 16 * Dsz;
    for (int k0 = 0; k0 < Dsz; k0 += 32) {
        __builtin_prefetch((const void*)(aBase0 + k0 + 128), 0, 1);
        __builtin_prefetch((const void*)(aBase1 + k0 + 128), 0, 1);
        FragBF a0, a1;
        a0.q[0] = *(const uint4*)(aBase0 + k0);
        a0.q[1] = *(const uint4*)(aBase0 + k0 + 16);
        a1.q[0] = *(const uint4*)(aBase1 + k0);
        a1.q[1] = *(const uint4*)(aBase1 + k0 + 16);
#pragma unroll
        for (int t = 0; t < 4; ++t) {
            const unsigned short* bp = wob + (size_t)(n0 + t * 16 + ln) * Dsz + k0 + kShift;
            __builtin_prefetch((const void*)(bp + 128), 0, 1);
            FragBF b;
            b.q[0] = *(const uint4*)(bp);
            b.q[1] = *(const uint4*)(bp + 16);
            acc[0][t].v = wmma_bf16(a0, b, acc[0][t].v);
            acc[1][t].v = wmma_bf16(a1, b, acc[1][t].v);
        }
    }
#pragma unroll
    for (int r = 0; r < 2; ++r) {
#pragma unroll
        for (int t = 0; t < 4; ++t) {
            const float bias = bo[n0 + t * 16 + ln];
#pragma unroll
            for (int mi = 0; mi < 8; ++mi) {
                out[(size_t)(m0 + r * 16 + mi + 8 * half) * Dsz + n0 + t * 16 + ln] =
                    acc[r][t].f[mi] + bias;
            }
        }
    }
}

// ---------------------------------------------------------------------------
// Host launch
// ---------------------------------------------------------------------------
extern "C" void kernel_launch(void* const* d_in, const int* in_sizes, int n_in,
                              void* d_out, int out_size, void* d_ws, size_t ws_size,
                              hipStream_t stream) {
    // inputs: x, mask, cos, sin, wq, bq, wk, wv, bv, wo, bo
    const float* x    = (const float*)d_in[0];
    const float* cosp = (const float*)d_in[2];
    const float* sinp = (const float*)d_in[3];
    const float* wq   = (const float*)d_in[4];
    const float* bq   = (const float*)d_in[5];
    const float* wk   = (const float*)d_in[6];
    const float* wv   = (const float*)d_in[7];
    const float* bv   = (const float*)d_in[8];
    const float* wo   = (const float*)d_in[9];
    const float* bo   = (const float*)d_in[10];

    const size_t NX = (size_t)Bsz * Ssz * Dsz;      // 4M
    const size_t NW = (size_t)Dsz * Dsz;            // 1M
    const size_t NH = (size_t)Bsz * Hsz * Ssz * HD; // 4M

    unsigned short* ws   = (unsigned short*)d_ws;
    unsigned short* xb   = ws;
    unsigned short* wqb  = xb   + NX;
    unsigned short* wkb  = wqb  + NW;
    unsigned short* wvb  = wkb  + NW;
    unsigned short* wob  = wvb  + NW;
    unsigned short* Qb   = wob  + NW;
    unsigned short* Kb   = Qb   + NH;
    unsigned short* Vtb  = Kb   + NH;
    unsigned short* vecb = Vtb  + NH;

    cvt_bf16_kernel<<<(int)((NX + 255) / 256), 256, 0, stream>>>(x,  xb,  (int)NX);
    cvt_bf16_kernel<<<(int)((NW + 255) / 256), 256, 0, stream>>>(wq, wqb, (int)NW);
    cvt_bf16_kernel<<<(int)((NW + 255) / 256), 256, 0, stream>>>(wk, wkb, (int)NW);
    cvt_bf16_kernel<<<(int)((NW + 255) / 256), 256, 0, stream>>>(wv, wvb, (int)NW);
    cvt_bf16_kernel<<<(int)((NW + 255) / 256), 256, 0, stream>>>(wo, wob, (int)NW);

    qkv_rope_kernel<<<dim3(Bsz * Ssz / 32, Hsz, 3), 32, 0, stream>>>(
        xb, wqb, wkb, wvb, bq, bv, cosp, sinp, Qb, Kb, Vtb);

    attn_kernel<<<dim3(Bsz * Hsz, Ssz / 16), 32, 0, stream>>>(Qb, Kb, Vtb, vecb);

    oproj_kernel<<<dim3(Bsz * Ssz / 32, Dsz / 64), 32, 0, stream>>>(
        vecb, wob, bo, (float*)d_out);
}
